// Decoder_77790447665279
// MI455X (gfx1250) — matrix operand order repaired and checked by reading, not compiled, not run on hardware
//
#include <hip/hip_runtime.h>
#include <hip/hip_bf16.h>
#include <math.h>

// Problem constants (from reference setup): B=16, T=128, H=512, L=2, V=32000
#define DB 16
#define DT 128
#define DH 512
#define DL 2
#define DV 32000
#define DM (DB * DT)      // 2048 rows of X
#define DG (4 * DH)       // 2048 gate columns

// GEMM tiling
#define BM 64
#define BN 64
#define BK 32
#define LDK 40            // padded LDS row stride (bf16 elems): 80B, bank-conflict-free

typedef __bf16 bf16_t;
typedef __attribute__((ext_vector_type(16))) __bf16 bf16x16;
typedef __attribute__((ext_vector_type(8)))  float  floatx8;

union Frag { bf16x16 v; uint4 q[2]; };

// ---------------------------------------------------------------------------
// f32 -> bf16 conversion (weights)
// ---------------------------------------------------------------------------
__global__ __launch_bounds__(256)
void f32_to_bf16_kernel(const float* __restrict__ src, bf16_t* __restrict__ dst, size_t n) {
    size_t i = (size_t)blockIdx.x * blockDim.x + threadIdx.x;
    if (i < n) dst[i] = (bf16_t)src[i];
}

// ---------------------------------------------------------------------------
// x0 = relu(embedding[tokens]) as bf16.  tokens = [BOS, target[:, :-1]]
// ---------------------------------------------------------------------------
__global__ __launch_bounds__(256)
void embed_relu_kernel(const int* __restrict__ target, const float* __restrict__ emb,
                       bf16_t* __restrict__ x0) {
    size_t idx = (size_t)blockIdx.x * blockDim.x + threadIdx.x;   // over DM*DH
    if (idx >= (size_t)DM * DH) return;
    int hh = (int)(idx % DH);
    int m  = (int)(idx / DH);
    int t  = m % DT;
    int b  = m / DT;
    int tok = (t == 0) ? 1 : target[b * DT + t - 1];   // BOS_IDX = 1
    float v = emb[(size_t)tok * DH + hh];
    x0[idx] = (bf16_t)(v > 0.f ? v : 0.f);
}

// ---------------------------------------------------------------------------
// gbias[l][b][g] = h0[l][b] . W_hh[l][g] + b_ih[l][g] + b_hh[l][g]
// ---------------------------------------------------------------------------
__global__ __launch_bounds__(256)
void gate_bias_kernel(const float* __restrict__ enc_h, const float* __restrict__ W_hh,
                      const float* __restrict__ b_ih, const float* __restrict__ b_hh,
                      float* __restrict__ gbias) {
    int idx = blockIdx.x * blockDim.x + threadIdx.x;   // over DL*DB*DG
    if (idx >= DL * DB * DG) return;
    int g = idx % DG;
    int b = (idx / DG) % DB;
    int l = idx / (DG * DB);
    const float* h0 = enc_h + ((size_t)l * DB + b) * DH;
    const float* w  = W_hh + ((size_t)l * DG + g) * DH;
    float s = 0.f;
#pragma unroll 8
    for (int k = 0; k < DH; ++k) s += h0[k] * w[k];
    gbias[idx] = s + b_ih[l * DG + g] + b_hh[l * DG + g];
}

// ---------------------------------------------------------------------------
// bf16 WMMA GEMM: out[m,n] = sum_k A[m,k] * W[n,k] + bias
//   MODE 0: bias[(m>>7)*N + n] (per-batch gate bias), regular stores
//   MODE 1: bias[n] (b_out), non-temporal stores (262MB stream > 192MB L2)
//
// Block = 128 threads (4 waves stacked in M). Block tile 64(M) x 64(N).
// K-slabs of 32 double-buffered through LDS (padded stride). Steady-state
// loop is branch-free. Inside compute(), the 4 B fragments live in DISTINCT
// registers and loads are interleaved between the WMMAs (prefetch distance
// 2), so each WMMA waits only on partial dscnt, not a full LDS round-trip.
// ---------------------------------------------------------------------------
template <int MODE>
__global__ __launch_bounds__(128)
void wmma_gemm_bf16(const bf16_t* __restrict__ A, const bf16_t* __restrict__ W,
                    const float* __restrict__ bias, float* __restrict__ out,
                    int M, int N, int K) {
    __shared__ bf16_t lA[2][BM * LDK];   // 2 x 5 KB
    __shared__ bf16_t lB[2][BN * LDK];   // 2 x 5 KB

    const int tid   = threadIdx.x;
    const int lane  = tid & 31;
    const int wave  = tid >> 5;
    const int nbase = blockIdx.x * BN;
    const int mbase = blockIdx.y * BM;
    const int lm = lane & 15;     // row (A) / col (B) within 16x16 tile
    const int kh = lane >> 4;     // K-half selector

    // Cooperative global->LDS loaders: thread covers 16 bf16 (32B) of one row
    // of A and one row of W per K-slab.
    const int grow  = tid >> 1;   // 0..63
    const int ghalf = tid & 1;    // which 16-elem half of the 32-wide slab
    const bf16_t* gA = A + (size_t)(mbase + grow) * K + ghalf * 16;
    const bf16_t* gW = W + (size_t)(nbase + grow) * K + ghalf * 16;

    uint4 ra0, ra1, rb0, rb1;
    auto gload = [&](int kk) {
        const uint4* pa = reinterpret_cast<const uint4*>(gA + kk);
        ra0 = pa[0]; ra1 = pa[1];
        const uint4* pb = reinterpret_cast<const uint4*>(gW + kk);
        rb0 = pb[0]; rb1 = pb[1];
    };
    auto sstore = [&](int buf) {
        uint4* da = reinterpret_cast<uint4*>(&lA[buf][grow * LDK + ghalf * 16]);
        da[0] = ra0; da[1] = ra1;
        uint4* db = reinterpret_cast<uint4*>(&lB[buf][grow * LDK + ghalf * 16]);
        db[0] = rb0; db[1] = rb1;
    };

    floatx8 acc[4] = {};
    auto ldB = [&](Frag& f, int buf, int j) {
        const bf16_t* bp = &lB[buf][(j * 16 + lm) * LDK + kh * 8];
        f.q[0] = *reinterpret_cast<const uint4*>(bp);
        f.q[1] = *reinterpret_cast<const uint4*>(bp + 16);
    };
    auto compute = [&](int buf) {
        Frag af, b0, b1, b2, b3;
        const bf16_t* ap = &lA[buf][(wave * 16 + lm) * LDK + kh * 8];
        af.q[0] = *reinterpret_cast<const uint4*>(ap);
        af.q[1] = *reinterpret_cast<const uint4*>(ap + 16);
        // Distinct live fragments + interleave: each WMMA's operand was
        // loaded >=2 load-pairs earlier -> partial dscnt waits only.
        ldB(b0, buf, 0);
        ldB(b1, buf, 1);
        acc[0] = __builtin_amdgcn_wmma_f32_16x16x32_bf16(
            false, af.v, false, b0.v, (short)0, acc[0], false, false);
        ldB(b2, buf, 2);
        acc[1] = __builtin_amdgcn_wmma_f32_16x16x32_bf16(
            false, af.v, false, b1.v, (short)0, acc[1], false, false);
        ldB(b3, buf, 3);
        acc[2] = __builtin_amdgcn_wmma_f32_16x16x32_bf16(
            false, af.v, false, b2.v, (short)0, acc[2], false, false);
        acc[3] = __builtin_amdgcn_wmma_f32_16x16x32_bf16(
            false, af.v, false, b3.v, (short)0, acc[3], false, false);
    };

    const int nsteps = K / BK;    // 16
    gload(0);
    sstore(0);
    __syncthreads();
    for (int s = 0; s < nsteps - 1; ++s) {
        gload((s + 1) * BK);      // HBM loads in flight during WMMAs
        compute(s & 1);
        sstore((s + 1) & 1);      // other buffer: safe before barrier
        __syncthreads();
    }
    compute((nsteps - 1) & 1);    // peeled final step: branch-free main loop

    // ---- epilogue: D elem r -> row (lane>=16?8:0)+r, col lane&15 ----
    const int mrow0 = mbase + wave * 16 + kh * 8;
#pragma unroll
    for (int j = 0; j < 4; ++j) {
        const int n = nbase + j * 16 + lm;
#pragma unroll
        for (int r = 0; r < 8; ++r) {
            const int m = mrow0 + r;
            float bv = (MODE == 0) ? bias[(size_t)(m >> 7) * N + n] : bias[n];
            float v = acc[j][r] + bv;
            if (MODE == 1) {
                __builtin_nontemporal_store(v, &out[(size_t)m * N + n]);
            } else {
                out[(size_t)m * N + n] = v;
            }
        }
    }
}

// ---------------------------------------------------------------------------
// LSTM cell activation epilogue for one layer:
//   c = sig(f)*c0[b] + sig(i)*tanh(g);  h = sig(o)*tanh(c)
// ---------------------------------------------------------------------------
__device__ __forceinline__ float sigm(float x) { return 1.f / (1.f + __expf(-x)); }

__global__ __launch_bounds__(256)
void lstm_act_kernel(const float* __restrict__ gates, const float* __restrict__ c0,
                     bf16_t* __restrict__ xout,
                     float* __restrict__ hlast, float* __restrict__ clast) {
    size_t idx = (size_t)blockIdx.x * blockDim.x + threadIdx.x;   // over DM*DH
    if (idx >= (size_t)DM * DH) return;
    int hh = (int)(idx % DH);
    int m  = (int)(idx / DH);
    int t  = m % DT;
    int b  = m / DT;
    const float* g = gates + (size_t)m * DG;
    float gi = g[hh];
    float gf = g[DH + hh];
    float gg = g[2 * DH + hh];
    float go = g[3 * DH + hh];
    float c = sigm(gf) * c0[b * DH + hh] + sigm(gi) * tanhf(gg);
    float h = sigm(go) * tanhf(c);
    xout[idx] = (bf16_t)h;
    if (t == DT - 1) {
        hlast[b * DH + hh] = h;
        clast[b * DH + hh] = c;
    }
}

// ---------------------------------------------------------------------------
// Host launcher
// ---------------------------------------------------------------------------
extern "C" void kernel_launch(void* const* d_in, const int* in_sizes, int n_in,
                              void* d_out, int out_size, void* d_ws, size_t ws_size,
                              hipStream_t stream) {
    (void)in_sizes; (void)n_in; (void)out_size; (void)ws_size;

    const float* enc_h  = (const float*)d_in[1];   // [L,B,H]
    const float* enc_c  = (const float*)d_in[2];   // [L,B,H]
    const int*   target = (const int*)  d_in[3];   // [B,T]
    const float* emb    = (const float*)d_in[4];   // [V,H]
    const float* W_ih   = (const float*)d_in[5];   // [L,4H,H]
    const float* W_hh   = (const float*)d_in[6];   // [L,4H,H]
    const float* b_ih   = (const float*)d_in[7];   // [L,4H]
    const float* b_hh   = (const float*)d_in[8];   // [L,4H]
    const float* W_out  = (const float*)d_in[9];   // [V,H]
    const float* b_out  = (const float*)d_in[10];  // [V]
    float* out = (float*)d_out;

    // Workspace carve-up (all offsets well aligned)
    char* ws = (char*)d_ws;
    size_t o = 0;
    bf16_t* x0      = (bf16_t*)(ws + o); o += (size_t)DM * DH * 2;            // 2 MB
    bf16_t* x1      = (bf16_t*)(ws + o); o += (size_t)DM * DH * 2;            // 2 MB
    bf16_t* wih_bf  = (bf16_t*)(ws + o); o += (size_t)DL * DG * DH * 2;       // 4 MB
    bf16_t* wout_bf = (bf16_t*)(ws + o); o += (size_t)DV * DH * 2;            // 32.8 MB
    float*  gbias   = (float*)(ws + o);  o += (size_t)DL * DB * DG * 4;       // 256 KB
    float*  gates   = (float*)(ws + o);  o += (size_t)DM * DG * 4;            // 16 MB

    // Output tail: logits [B,T,V], then h_stack [L,B,H], then c_stack [L,B,H]
    const size_t LOGITS = (size_t)DB * DT * DV;
    float* hstack = out + LOGITS;
    float* cstack = out + LOGITS + (size_t)DL * DB * DH;

    // 1) Convert weights to bf16
    {
        size_t n = (size_t)DL * DG * DH;                    // 2,097,152
        f32_to_bf16_kernel<<<(int)((n + 255) / 256), 256, 0, stream>>>(W_ih, wih_bf, n);
        size_t n2 = (size_t)DV * DH;                        // 16,384,000
        f32_to_bf16_kernel<<<(int)((n2 + 255) / 256), 256, 0, stream>>>(W_out, wout_bf, n2);
    }

    // 2) Embedding gather + ReLU -> x0 (bf16)
    embed_relu_kernel<<<(DM * DH + 255) / 256, 256, 0, stream>>>(target, emb, x0);

    // 3) Per-batch gate bias: h0 @ W_hh^T + b_ih + b_hh  (both layers)
    gate_bias_kernel<<<(DL * DB * DG + 255) / 256, 256, 0, stream>>>(enc_h, W_hh, b_ih, b_hh, gbias);

    // 4) Layer 0: gates = x0 @ W_ih[0]^T + gbias[0]; activations -> x1
    {
        dim3 grid(DG / BN, DM / BM);
        wmma_gemm_bf16<0><<<grid, 128, 0, stream>>>(x0, wih_bf, gbias, gates, DM, DG, DH);
        lstm_act_kernel<<<(DM * DH + 255) / 256, 256, 0, stream>>>(
            gates, enc_c + 0, x1, hstack + 0, cstack + 0);
    }

    // 5) Layer 1: gates = x1 @ W_ih[1]^T + gbias[1]; activations -> x0 (reused as x2)
    {
        dim3 grid(DG / BN, DM / BM);
        wmma_gemm_bf16<0><<<grid, 128, 0, stream>>>(
            x1, wih_bf + (size_t)DG * DH, gbias + (size_t)DB * DG, gates, DM, DG, DH);
        lstm_act_kernel<<<(DM * DH + 255) / 256, 256, 0, stream>>>(
            gates, enc_c + (size_t)DB * DH, x0, hstack + (size_t)DB * DH, cstack + (size_t)DB * DH);
    }

    // 6) Logits: out = x2 @ W_out^T + b_out   (M=2048, N=32000, K=512)
    {
        dim3 grid(DV / BN, DM / BM);
        wmma_gemm_bf16<1><<<grid, 128, 0, stream>>>(x0, wout_bf, b_out, out, DM, DV, DH);
    }
}
